// ResidualVectorQuantizer_8100308320912
// MI455X (gfx1250) — matrix operand order repaired and checked by reading, not compiled, not run on hardware
//
#include <hip/hip_runtime.h>
#include <hip/hip_bf16.h>

typedef __attribute__((ext_vector_type(16))) _Float16 v16h;
typedef __attribute__((ext_vector_type(8)))  float    v8f;

#define RVQ_Q 8
#define RVQ_C 1024
#define RVQ_D 256
#define RVQ_N 32768   // 32 * 1024 rows

union Frag16 { v16h v; uint32_t u[8]; };

__device__ __forceinline__ float wave_red_sum(float s) {
  #pragma unroll
  for (int m = 16; m >= 1; m >>= 1) s += __shfl_xor(s, m, 32);
  return s;
}

// WGP-scope prefetch (scope field defaults to 0 = WGP: fill all cache levels).
__device__ __forceinline__ void prefetch_wgp(const void* p) {
  asm volatile("global_prefetch_b8 %0, off" : : "v"(p));
}
__device__ __forceinline__ void prefetch_wgp_off128(const void* p) {
  asm volatile("global_prefetch_b8 %0, off offset:128" : : "v"(p));
}

// ---------------- one-time: L2-normalize all codebooks to f16 ----------------
__global__ __launch_bounds__(256) void rvq_norm_cb(const float* __restrict__ cb,
                                                   _Float16* __restrict__ cbN) {
  const int row  = blockIdx.x * 8 + (threadIdx.x >> 5);   // one wave per row
  const int lane = threadIdx.x & 31;
  const float* rp = cb + (size_t)row * RVQ_D;
  float s = 0.f;
  #pragma unroll
  for (int i = 0; i < 8; ++i) { float x = rp[lane + i * 32]; s += x * x; }
  s = wave_red_sum(s);
  const float sc = 1.f / fmaxf(sqrtf(s), 1e-12f);
  #pragma unroll
  for (int i = 0; i < 8; ++i)
    cbN[(size_t)row * RVQ_D + lane + i * 32] = (_Float16)(rp[lane + i * 32] * sc);
}

// ---------------- init: residual = inputs, qsum = 0, loss = 0 ----------------
__global__ __launch_bounds__(256) void rvq_init(const float4* __restrict__ in,
                                                float4* __restrict__ resid,
                                                float4* __restrict__ qsum,
                                                float* __restrict__ lossAcc) {
  const int t = blockIdx.x * 256 + threadIdx.x;
  float4 v = in[t];
  resid[t] = v;
  qsum[t]  = make_float4(0.f, 0.f, 0.f, 0.f);
  if (t == 0) *lossAcc = 0.f;
}

// ---------- per stage: normalize residual rows, WMMA sim, row argmax ---------
// 4 waves/block; each wave owns M=32 rows (two 16-row A tiles) so every B
// fragment feeds two WMMAs; next column tile prefetched at WGP scope.
__global__ __launch_bounds__(128) void rvq_argmax(const float* __restrict__ resid,
                                                  const uint32_t* __restrict__ cbN,
                                                  int* __restrict__ idxOut,
                                                  float* __restrict__ idxOutF,
                                                  int stage) {
  __shared__ uint32_t sm[4 * 32 * 128];                  // 4 waves * 32 rows * 256 f16 = 64KB
  const int wave = threadIdx.x >> 5;
  const int lane = threadIdx.x & 31;
  const int r    = lane & 15;
  const int hi   = lane >> 4;                            // which 128-elem half
  const int rowBase = (blockIdx.x * 4 + wave) * 32;
  uint32_t* smw = sm + wave * (32 * 128);

  // --- normalize my 32 rows of residual into LDS as packed f16 ---
  #pragma unroll
  for (int t2 = 0; t2 < 32; t2 += 16) {
    const float4* rp = (const float4*)(resid + (size_t)(rowBase + t2 + r) * RVQ_D + hi * 128);
    float s = 0.f;
    #pragma unroll
    for (int i = 0; i < 32; ++i) { float4 x = rp[i]; s += x.x*x.x + x.y*x.y + x.z*x.z + x.w*x.w; }
    s += __shfl_xor(s, 16, 32);
    const float sc = 1.f / fmaxf(sqrtf(s), 1e-12f);
    #pragma unroll
    for (int i = 0; i < 32; ++i) {
      float4 x = rp[i];
      union { uint32_t u; _Float16 h[2]; } p0, p1;
      p0.h[0] = (_Float16)(x.x * sc); p0.h[1] = (_Float16)(x.y * sc);
      p1.h[0] = (_Float16)(x.z * sc); p1.h[1] = (_Float16)(x.w * sc);
      smw[(t2 + r) * 128 + hi * 64 + i * 2]     = p0.u;
      smw[(t2 + r) * 128 + hi * 64 + i * 2 + 1] = p1.u;
    }
  }
  __syncthreads();

  // --- prebuild A fragments for both 16-row tiles (contiguous k = 16*hi + s order, same as B) ---
  const int abase = (lane < 16) ? 0 : 16;
  v16h afr0[8], afr1[8];
  #pragma unroll
  for (int kc = 0; kc < 8; ++kc) {
    Frag16 A0, A1;
    #pragma unroll
    for (int j = 0; j < 8; ++j) {
      int k = kc * 32 + abase + (j << 1);
      A0.u[j] = smw[r * 128 + (k >> 1)];
      A1.u[j] = smw[(16 + r) * 128 + (k >> 1)];
    }
    afr0[kc] = A0.v;
    afr1[kc] = A1.v;
  }

  const uint32_t* cbS = cbN + (size_t)stage * RVQ_C * (RVQ_D / 2);
  const int kb2 = (lane < 16) ? 0 : 8;                   // B layout: K half select
  float bval0[8], bval1[8];
  int   bidx0[8], bidx1[8];
  #pragma unroll
  for (int j = 0; j < 8; ++j) {
    bval0[j] = -3.4e38f; bidx0[j] = 0;
    bval1[j] = -3.4e38f; bidx1[j] = 0;
  }

  #pragma unroll 1
  for (int ct = 0; ct < RVQ_C / 16; ++ct) {
    const int colBase = ct * 16;
    const uint32_t* bp = cbS + (size_t)(colBase + r) * (RVQ_D / 2) + kb2;

    // WGP-scope prefetch of next column tile (stays inside mapped workspace);
    // lanes x hi cover all four 128B lines of each 512B codebook row
    const char* nb = (const char*)(cbS + (size_t)(colBase + 16 + r) * (RVQ_D / 2)) + hi * 256;
    prefetch_wgp(nb);
    prefetch_wgp_off128(nb);

    v8f acc0 = {0.f, 0.f, 0.f, 0.f, 0.f, 0.f, 0.f, 0.f};
    v8f acc1 = {0.f, 0.f, 0.f, 0.f, 0.f, 0.f, 0.f, 0.f};
    #pragma unroll
    for (int kc = 0; kc < 8; ++kc) {
      Frag16 B;
      #pragma unroll
      for (int j = 0; j < 8; ++j) B.u[j] = bp[kc * 16 + j];
      acc0 = __builtin_amdgcn_wmma_f32_16x16x32_f16(false, afr0[kc], false, B.v,
                                                    (short)0, acc0, false, false);
      acc1 = __builtin_amdgcn_wmma_f32_16x16x32_f16(false, afr1[kc], false, B.v,
                                                    (short)0, acc1, false, false);
    }

    const int col = colBase + r;
    #pragma unroll
    for (int j = 0; j < 8; ++j) {                        // lane holds (M=j+8*hi, N=col)
      float v0 = acc0[j];
      if (v0 > bval0[j]) { bval0[j] = v0; bidx0[j] = col; }
      float v1 = acc1[j];
      if (v1 > bval1[j]) { bval1[j] = v1; bidx1[j] = col; }
    }
  }

  // --- merge across the 16 lanes holding the same row (masks stay in half) ---
  #pragma unroll
  for (int j = 0; j < 8; ++j) {
    #pragma unroll
    for (int m = 8; m >= 1; m >>= 1) {
      float ov0 = __shfl_xor(bval0[j], m, 32);
      int   oi0 = __shfl_xor(bidx0[j], m, 32);
      if (ov0 > bval0[j] || (ov0 == bval0[j] && oi0 < bidx0[j])) { bval0[j] = ov0; bidx0[j] = oi0; }
      float ov1 = __shfl_xor(bval1[j], m, 32);
      int   oi1 = __shfl_xor(bidx1[j], m, 32);
      if (ov1 > bval1[j] || (ov1 == bval1[j] && oi1 < bidx1[j])) { bval1[j] = ov1; bidx1[j] = oi1; }
    }
  }
  if (r == 0) {                                          // lanes 0 and 16
    const int mrow0 = rowBase + hi * 8;
    const int mrow1 = rowBase + 16 + hi * 8;
    #pragma unroll
    for (int j = 0; j < 8; ++j) {
      idxOut[mrow0 + j] = bidx0[j];
      idxOutF[(size_t)stage * RVQ_N + mrow0 + j] = (float)bidx0[j];
      idxOut[mrow1 + j] = bidx1[j];
      idxOutF[(size_t)stage * RVQ_N + mrow1 + j] = (float)bidx1[j];
    }
  }
}

// -------- per stage: gather code, update residual/qsum, accumulate loss ------
__global__ __launch_bounds__(256) void rvq_update(const float* __restrict__ cb,
                                                  const int* __restrict__ idx,
                                                  float* __restrict__ resid,
                                                  float* __restrict__ qsum,
                                                  float* __restrict__ lossAcc,
                                                  int stage) {
  const int t   = blockIdx.x * 256 + threadIdx.x;        // 0 .. N*D/4
  const int row = t >> 6;
  const int db  = (t & 63) << 2;
  const float4 q = *(const float4*)(cb + ((size_t)stage * RVQ_C + idx[row]) * RVQ_D + db);
  float4* rp = (float4*)resid + t;
  float4  rv = *rp;
  const float dx = q.x - rv.x, dy = q.y - rv.y, dz = q.z - rv.z, dw = q.w - rv.w;
  float4* qp = (float4*)qsum + t;
  float4  qs = *qp;
  qs.x += q.x; qs.y += q.y; qs.z += q.z; qs.w += q.w;
  *qp = qs;
  rv.x -= q.x; rv.y -= q.y; rv.z -= q.z; rv.w -= q.w;
  *rp = rv;
  float s = dx * dx + dy * dy + dz * dz + dw * dw;
  s = wave_red_sum(s);
  if ((threadIdx.x & 31) == 0) atomicAdd(lossAcc, s);
}

__global__ void rvq_finalize(const float* __restrict__ lossAcc, float* __restrict__ out) {
  out[0] = lossAcc[0] * (1.25f / (8.f * (float)RVQ_N * (float)RVQ_D));
}

extern "C" void kernel_launch(void* const* d_in, const int* in_sizes, int n_in,
                              void* d_out, int out_size, void* d_ws, size_t ws_size,
                              hipStream_t stream) {
  (void)in_sizes; (void)n_in; (void)out_size; (void)ws_size;
  const float* inp = (const float*)d_in[0];
  const float* cbs = (const float*)d_in[1];
  float* out = (float*)d_out;

  uint8_t* w = (uint8_t*)d_ws;
  float* resid = (float*)w;
  size_t off = (size_t)RVQ_N * RVQ_D * sizeof(float);               // 32 MB
  _Float16* cbN = (_Float16*)(w + off);
  off += (size_t)RVQ_Q * RVQ_C * RVQ_D * sizeof(_Float16);          // +4 MB
  int* idx = (int*)(w + off);
  off += (size_t)RVQ_N * sizeof(int);                               // +128 KB
  float* lossAcc = (float*)(w + off);

  float* qsum    = out;
  float* lossOut = out + (size_t)RVQ_N * RVQ_D;
  float* idxOutF = lossOut + 1;

  rvq_norm_cb<<<RVQ_Q * RVQ_C / 8, 256, 0, stream>>>(cbs, cbN);
  rvq_init<<<(RVQ_N * RVQ_D / 4) / 256, 256, 0, stream>>>(
      (const float4*)inp, (float4*)resid, (float4*)qsum, lossAcc);
  for (int q = 0; q < RVQ_Q; ++q) {
    rvq_argmax<<<RVQ_N / 128, 128, 0, stream>>>(resid, (const uint32_t*)cbN, idx, idxOutF, q);
    rvq_update<<<(RVQ_N * RVQ_D / 4) / 256, 256, 0, stream>>>(cbs, idx, resid, qsum, lossAcc, q);
  }
  rvq_finalize<<<1, 1, 0, stream>>>(lossAcc, lossOut);
}
